// PCDKNNGroupEncoder_68083821576810
// MI455X (gfx1250) — compile-verified, matrix operand order checked by reading, not run on hardware
//
#include <hip/hip_runtime.h>
#include <hip/hip_bf16.h>

// ---------------------------------------------------------------------------
// PCD KNN Group Encoder for gfx1250 (MI455X), wave32 + WMMA f16 16x16x32.
// Pipeline: prep(fold BN, transpose weights to f16 col-major) -> FPS ->
//           KNN+gather (f16 padded 32x32 tiles) -> per-group WMMA MLP with
//           async global->LDS staging of the A tile (ASYNCcnt path).
// ---------------------------------------------------------------------------

typedef _Float16 half8_t  __attribute__((ext_vector_type(8)));
typedef _Float16 half16_t __attribute__((ext_vector_type(16)));
typedef float    f32x8    __attribute__((ext_vector_type(8)));

#define NPTS   16384
#define NGRP   64
#define KNN    32
#define NBATCH 32
#define NGROUPS_TOTAL (NBATCH * NGRP)   // 2048

// ---------------------------------------------------------------------------
// Fragment loader matching the CDNA5 16-bit A-matrix 16x32 layout (CDNA5 ISA):
// lane l (r = l&15, hi = l>>4) holds K = hi*8 + 0..7 in v0..v3 and
// K = 16 + hi*8 + 0..7 in v4..v7 -> two contiguous 16-byte runs per lane.
// Used for A (row-major [row][k]) and B (weights pre-transposed to [n][k]).
// ---------------------------------------------------------------------------
__device__ __forceinline__ half16_t load_frag(const _Float16* base, int rowbase,
                                              int stride, int kbase, int lane) {
  const int r  = rowbase + (lane & 15);
  const int hi = (lane >> 4) & 1;
  const _Float16* p = base + (size_t)r * stride + kbase + hi * 8;
  half8_t lo = *(const half8_t*)(p);
  half8_t up = *(const half8_t*)(p + 16);
  return __builtin_shufflevector(lo, up, 0,1,2,3,4,5,6,7,8,9,10,11,12,13,14,15);
}

// D-tile (16x16 f32) rows per lane: lanes 0-15 -> M=i, lanes 16-31 -> M=i+8.
__device__ __forceinline__ void gemm_stage(const _Float16* A, int astride,
                                           const _Float16* WT, int kdim,
                                           const float* bias, bool relu,
                                           _Float16* outbuf, int ostride, int ocolofs,
                                           int ntiles, int wave, int lane) {
  const int ksteps = kdim >> 5;
  for (int t = wave; t < 2 * ntiles; t += 8) {
    const int mt = t & 1, nt = t >> 1;
    f32x8 acc = {};
    for (int kk = 0; kk < ksteps; ++kk) {
      if (kk + 1 < ksteps)
        __builtin_prefetch(WT + (size_t)(nt * 16 + (lane & 15)) * kdim + (kk + 1) * 32, 0, 0);
      half16_t a = load_frag(A,  mt * 16, astride, kk * 32, lane);
      half16_t b = load_frag(WT, nt * 16, kdim,    kk * 32, lane);
      acc = __builtin_amdgcn_wmma_f32_16x16x32_f16(false, a, false, b,
                                                   (short)0, acc, false, false);
    }
    const int col = nt * 16 + (lane & 15);
    const int hi  = (lane >> 4) & 1;
    const float bb = bias[col];
#pragma unroll
    for (int i = 0; i < 8; ++i) {
      float v = acc[i] + bb;
      if (relu) v = fmaxf(v, 0.0f);
      outbuf[(size_t)(mt * 16 + hi * 8 + i) * ostride + ocolofs + col] = (_Float16)v;
    }
  }
}

// Same, but f32 output and no bias/ReLU (bias folded in after the K-max).
__device__ __forceinline__ void gemm_stage_f32(const _Float16* A, int astride,
                                               const _Float16* WT, int kdim,
                                               float* outbuf, int ostride,
                                               int ntiles, int wave, int lane) {
  const int ksteps = kdim >> 5;
  for (int t = wave; t < 2 * ntiles; t += 8) {
    const int mt = t & 1, nt = t >> 1;
    f32x8 acc = {};
    for (int kk = 0; kk < ksteps; ++kk) {
      half16_t a = load_frag(A,  mt * 16, astride, kk * 32, lane);
      half16_t b = load_frag(WT, nt * 16, kdim,    kk * 32, lane);
      acc = __builtin_amdgcn_wmma_f32_16x16x32_f16(false, a, false, b,
                                                   (short)0, acc, false, false);
    }
    const int col = nt * 16 + (lane & 15);
    const int hi  = (lane >> 4) & 1;
#pragma unroll
    for (int i = 0; i < 8; ++i)
      outbuf[(size_t)(mt * 16 + hi * 8 + i) * ostride + col] = acc[i];
  }
}

// ---------------------------------------------------------------------------
// Kernel 1: fold BN into weights, transpose to col-major [n][k], convert f16.
// ---------------------------------------------------------------------------
__global__ void prep_weights_kernel(
    const float* __restrict__ w1, const float* __restrict__ b1,
    const float* __restrict__ g1, const float* __restrict__ be1,
    const float* __restrict__ m1, const float* __restrict__ v1,
    const float* __restrict__ w2, const float* __restrict__ b2,
    const float* __restrict__ w3, const float* __restrict__ b3,
    const float* __restrict__ g2, const float* __restrict__ be2,
    const float* __restrict__ m2, const float* __restrict__ v2,
    const float* __restrict__ w4, const float* __restrict__ b4,
    _Float16* __restrict__ W1T, _Float16* __restrict__ W2T,
    _Float16* __restrict__ W3T, _Float16* __restrict__ W4T,
    float* __restrict__ b1f, float* __restrict__ b2f,
    float* __restrict__ b3f, float* __restrict__ b4f) {
  const int i = blockIdx.x * 256 + threadIdx.x;
  if (i < 4096) {                               // w1: (6,128) -> W1T[128][32] pad
    const int d = i >> 5, k = i & 31;
    const float s = g1[d] * rsqrtf(v1[d] + 1e-5f);
    W1T[d * 32 + k] = (_Float16)((k < 6) ? w1[k * 128 + d] * s : 0.0f);
    if (k == 0) b1f[d] = (b1[d] - m1[d]) * s + be1[d];
  } else if (i < 4096 + 32768) {                // w2: (128,256) -> W2T[256][128]
    const int j = i - 4096, d = j >> 7, k = j & 127;
    W2T[d * 128 + k] = (_Float16)w2[k * 256 + d];
    if (k == 0) b2f[d] = b2[d];
  } else if (i < 4096 + 32768 + 262144) {       // w3: (512,512) -> W3T[512][512]
    const int j = i - 36864, d = j >> 9, k = j & 511;
    const float s = g2[d] * rsqrtf(v2[d] + 1e-5f);
    W3T[d * 512 + k] = (_Float16)(w3[k * 512 + d] * s);
    if (k == 0) b3f[d] = (b3[d] - m2[d]) * s + be2[d];
  } else if (i < 495616) {                      // w4: (512,384) -> W4T[384][512]
    const int j = i - 299008, d = j >> 9, k = j & 511;
    W4T[d * 512 + k] = (_Float16)w4[k * 384 + d];
    if (k == 0) b4f[d] = b4[d];
  }
}

// ---------------------------------------------------------------------------
// Kernel 2: farthest point sampling, 1 block per batch, dists live in LDS.
// argmax via packed (float-bits << 32 | ~idx) u64 max tree-reduction.
// ---------------------------------------------------------------------------
__global__ void fps_kernel(const float* __restrict__ pcd,
                           int* __restrict__ cidx, float* __restrict__ centers) {
  extern __shared__ __align__(16) char smem_fps[];
  float* dists = (float*)smem_fps;                                  // 16384 f32
  unsigned long long* keys = (unsigned long long*)(smem_fps + NPTS * 4);  // 256
  float* ctr = (float*)(smem_fps + NPTS * 4 + 256 * 8);             // 3 f32

  const int b   = blockIdx.x;
  const int tid = threadIdx.x;
  const float* px = pcd + (size_t)b * 6 * NPTS;
  const float* py = px + NPTS;
  const float* pz = px + 2 * NPTS;

  for (int i = tid; i < NPTS; i += 256) dists[i] = 1e10f;
  __syncthreads();

  int far = 0;
  for (int g = 0; g < NGRP; ++g) {
    if (tid == 0) {
      cidx[b * NGRP + g] = far;
      const float cx = px[far], cy = py[far], cz = pz[far];
      ctr[0] = cx; ctr[1] = cy; ctr[2] = cz;
      float* cc = centers + (size_t)(b * NGRP + g) * 3;
      cc[0] = cx; cc[1] = cy; cc[2] = cz;
    }
    __syncthreads();
    const float cx = ctr[0], cy = ctr[1], cz = ctr[2];
    unsigned long long best = 0ull;
    for (int i = tid; i < NPTS; i += 256) {
      const float dx = px[i] - cx, dy = py[i] - cy, dz = pz[i] - cz;
      float d = dx * dx + dy * dy + dz * dz;
      const float old = dists[i];
      d = (d < old) ? d : old;
      dists[i] = d;
      const unsigned long long k =
          ((unsigned long long)__float_as_uint(d) << 32) | (unsigned)(NPTS - 1 - i);
      best = (k > best) ? k : best;
    }
    keys[tid] = best;
    __syncthreads();
    for (int s = 128; s > 0; s >>= 1) {
      if (tid < s) {
        const unsigned long long o = keys[tid + s];
        if (o > keys[tid]) keys[tid] = o;
      }
      __syncthreads();
    }
    far = NPTS - 1 - (int)(unsigned)(keys[0] & 0xFFFFFFFFull);
    __syncthreads();
  }
}

// ---------------------------------------------------------------------------
// Kernel 3: KNN-32 + gather. 1 block per (batch, group). d2 array in LDS,
// 32 rounds of packed u64 min-reduction; neighbor order is irrelevant
// downstream (all K reductions are max), so any tie order is fine.
// Emits the f16, zero-padded 32x32 A-tile for the WMMA MLP.
// ---------------------------------------------------------------------------
__global__ void knn_gather_kernel(const float* __restrict__ pcd,
                                  const float* __restrict__ centers,
                                  _Float16* __restrict__ gA) {
  extern __shared__ __align__(16) char smem_knn[];
  float* d2 = (float*)smem_knn;                                     // 16384 f32
  unsigned long long* keys = (unsigned long long*)(smem_knn + NPTS * 4);  // 256
  int* sel = (int*)(smem_knn + NPTS * 4 + 256 * 8);                 // 32 ints

  const int n   = blockIdx.x;            // n = b*64 + g
  const int b   = n >> 6;
  const int tid = threadIdx.x;
  const float* px = pcd + (size_t)b * 6 * NPTS;
  const float* py = px + NPTS;
  const float* pz = px + 2 * NPTS;
  const float cx = centers[(size_t)n * 3 + 0];
  const float cy = centers[(size_t)n * 3 + 1];
  const float cz = centers[(size_t)n * 3 + 2];

  for (int i = tid; i < NPTS; i += 256) {
    const float dx = px[i] - cx, dy = py[i] - cy, dz = pz[i] - cz;
    d2[i] = dx * dx + dy * dy + dz * dz;
  }
  __syncthreads();

  for (int r = 0; r < KNN; ++r) {
    unsigned long long best = ~0ull;
    for (int i = tid; i < NPTS; i += 256) {
      const unsigned long long k =
          ((unsigned long long)__float_as_uint(d2[i]) << 32) | (unsigned)i;
      best = (k < best) ? k : best;
    }
    keys[tid] = best;
    __syncthreads();
    for (int s = 128; s > 0; s >>= 1) {
      if (tid < s) {
        const unsigned long long o = keys[tid + s];
        if (o < keys[tid]) keys[tid] = o;
      }
      __syncthreads();
    }
    if (tid == 0) {
      const int idx = (int)(unsigned)(keys[0] & 0xFFFFFFFFull);
      sel[r] = idx;
      d2[idx] = 3.4e38f;
    }
    __syncthreads();
  }

  _Float16* tile = gA + (size_t)n * (KNN * 32);
  for (int t = tid; t < KNN * 32; t += 256) tile[t] = (_Float16)0.0f;
  __syncthreads();
  if (tid < KNN * 6) {
    const int k = tid / 6, c = tid % 6;
    float v = pcd[(size_t)b * 6 * NPTS + (size_t)c * NPTS + sel[k]];
    if (c == 0) v -= cx;
    if (c == 1) v -= cy;
    if (c == 2) v -= cz;
    tile[k * 32 + c] = (_Float16)v;
  }
}

// ---------------------------------------------------------------------------
// Kernel 4: per-group MLP, 8 wave32s per block, all GEMMs on v_wmma.
// The 2 KB A-tile is staged into LDS once through the CDNA5 async
// global->LDS engine (ASYNCcnt) instead of being re-read 16x from L2.
// LDS: sF1 32x128 f16 | sCat 32x512 f16 ([fg|f2]) | sF3 32x512 f16 |
//      sF4 32x384 f32 | sA 32x32 f16  => 124928 B dynamic.
// ---------------------------------------------------------------------------
__global__ void mlp_kernel(const _Float16* __restrict__ gA,
                           const _Float16* __restrict__ W1T,
                           const _Float16* __restrict__ W2T,
                           const _Float16* __restrict__ W3T,
                           const _Float16* __restrict__ W4T,
                           const float* __restrict__ b1f,
                           const float* __restrict__ b2f,
                           const float* __restrict__ b3f,
                           const float* __restrict__ b4f,
                           float* __restrict__ out) {
  extern __shared__ __align__(16) char smem_mlp[];
  _Float16* sF1  = (_Float16*)(smem_mlp);                         // 8192 B
  _Float16* sCat = (_Float16*)(smem_mlp + 8192);                  // 32768 B
  _Float16* sF3  = (_Float16*)(smem_mlp + 8192 + 32768);          // 32768 B
  float*    sF4  = (float*)   (smem_mlp + 8192 + 65536);          // 49152 B
  _Float16* sA   = (_Float16*)(smem_mlp + 8192 + 65536 + 49152);  // 2048 B

  const int n    = blockIdx.x;
  const int tid  = threadIdx.x;
  const int wave = tid >> 5;
  const int lane = tid & 31;
  const _Float16* A0 = gA + (size_t)n * (KNN * 32);

  // Async-stage the 2 KB A-tile: 128 lanes x b128. The LDS VDST operand is
  // the low 32 bits of the generic pointer (ISA 10.2: LDS aperture keeps the
  // wave-relative offset in addr[31:0]).
  if (tid < 128) {
    const unsigned lds_dst =
        (unsigned)(unsigned long long)(void*)sA + (unsigned)tid * 16u;
    const unsigned long long gsrc =
        (unsigned long long)(const void*)A0 + (unsigned long long)tid * 16ull;
    asm volatile("global_load_async_to_lds_b128 %0, %1, off"
                 :: "v"(lds_dst), "v"(gsrc) : "memory");
  }
  asm volatile("s_wait_asynccnt 0x0" ::: "memory");
  __syncthreads();

  // stage 1: (32x32 pad, from LDS) @ W1T -> relu(bn-folded) -> sF1 (32x128)
  gemm_stage(sA, 32, W1T, 32, b1f, true, sF1, 128, 0, 8, wave, lane);
  __syncthreads();
  // stage 2: sF1 @ W2T -> sCat[:,256:512] (32x256)
  gemm_stage(sF1, 128, W2T, 128, b2f, false, sCat, 512, 256, 16, wave, lane);
  __syncthreads();
  // fg = max over K rows of f2, broadcast into sCat[:,0:256]
  {
    const int col = tid;  // 256 threads, one column each
    float m = (float)sCat[256 + col];
    for (int r = 1; r < KNN; ++r) m = fmaxf(m, (float)sCat[r * 512 + 256 + col]);
    const _Float16 mh = (_Float16)m;
    for (int r = 0; r < KNN; ++r) sCat[r * 512 + col] = mh;
  }
  __syncthreads();
  // stage 3: sCat(32x512) @ W3T -> relu(bn-folded) -> sF3 (32x512)
  gemm_stage(sCat, 512, W3T, 512, b3f, true, sF3, 512, 0, 32, wave, lane);
  __syncthreads();
  // stage 4: sF3 @ W4T -> sF4 (32x384 f32), bias deferred past the max
  gemm_stage_f32(sF3, 512, W4T, 512, sF4, 384, 24, wave, lane);
  __syncthreads();
  // out[n, c] = max_k f4[k, c] + b4[c]
  for (int col = tid; col < 384; col += 256) {
    float m = sF4[col];
    for (int r = 1; r < KNN; ++r) m = fmaxf(m, sF4[r * 384 + col]);
    out[(size_t)n * 384 + col] = m + b4f[col];
  }
}

// ---------------------------------------------------------------------------
extern "C" void kernel_launch(void* const* d_in, const int* in_sizes, int n_in,
                              void* d_out, int out_size, void* d_ws, size_t ws_size,
                              hipStream_t stream) {
  (void)in_sizes; (void)n_in; (void)out_size; (void)ws_size;
  const float* pcd = (const float*)d_in[0];
  // d_in[1] = pcd_mask (unused by reference math)
  const float* w1  = (const float*)d_in[2];
  const float* b1  = (const float*)d_in[3];
  const float* g1  = (const float*)d_in[4];
  const float* be1 = (const float*)d_in[5];
  const float* m1  = (const float*)d_in[6];
  const float* v1  = (const float*)d_in[7];
  const float* w2  = (const float*)d_in[8];
  const float* b2  = (const float*)d_in[9];
  const float* w3  = (const float*)d_in[10];
  const float* b3  = (const float*)d_in[11];
  const float* g2  = (const float*)d_in[12];
  const float* be2 = (const float*)d_in[13];
  const float* m2  = (const float*)d_in[14];
  const float* v2  = (const float*)d_in[15];
  const float* w4  = (const float*)d_in[16];
  const float* b4  = (const float*)d_in[17];

  char* ws = (char*)d_ws;
  size_t off = 0;
  _Float16* W1T = (_Float16*)(ws + off); off += 8192;     // [128][32]
  _Float16* W2T = (_Float16*)(ws + off); off += 65536;    // [256][128]
  _Float16* W3T = (_Float16*)(ws + off); off += 524288;   // [512][512]
  _Float16* W4T = (_Float16*)(ws + off); off += 393216;   // [384][512]
  float* b1f = (float*)(ws + off); off += 512;
  float* b2f = (float*)(ws + off); off += 1024;
  float* b3f = (float*)(ws + off); off += 2048;
  float* b4f = (float*)(ws + off); off += 1536;
  int*   cidx    = (int*)(ws + off);   off += 8192;       // [32][64]
  float* centers = (float*)(ws + off); off += 24576;      // [2048][3]
  _Float16* gA = (_Float16*)(ws + off); off += 4194304;   // [2048][32][32]

  prep_weights_kernel<<<1936, 256, 0, stream>>>(
      w1, b1, g1, be1, m1, v1, w2, b2, w3, b3, g2, be2, m2, v2, w4, b4,
      W1T, W2T, W3T, W4T, b1f, b2f, b3f, b4f);
  fps_kernel<<<NBATCH, 256, 67712, stream>>>(pcd, cidx, centers);
  knn_gather_kernel<<<NGROUPS_TOTAL, 256, 67712, stream>>>(pcd, centers, gA);
  mlp_kernel<<<NGROUPS_TOTAL, 256, 124928, stream>>>(
      gA, W1T, W2T, W3T, W4T, b1f, b2f, b3f, b4f, (float*)d_out);
}
